// MultiHeadAttention_2963527434336
// MI455X (gfx1250) — matrix operand order repaired and checked by reading, not compiled, not run on hardware
//
#include <hip/hip_runtime.h>
#include <cstdint>

// ---------------- optional CDNA5 async-to-LDS path ----------------
#if defined(__has_builtin)
#  if __has_builtin(__builtin_amdgcn_global_load_async_to_lds_b128) && \
      __has_builtin(__builtin_amdgcn_s_wait_asynccnt)
#    define USE_ASYNC_LDS 1
#  endif
#endif
#ifndef USE_ASYNC_LDS
#  define USE_ASYNC_LDS 0
#endif

#if USE_ASYNC_LDS
typedef int v4i32_t __attribute__((ext_vector_type(4)));
typedef __attribute__((address_space(1))) v4i32_t* gbl_v4i_ptr;   // global
typedef __attribute__((address_space(3))) v4i32_t* lds_v4i_ptr;   // LDS
__device__ __forceinline__ void async_copy_b128(const void* g, void* l) {
  // integer casts sidestep address-space pointer-cast restrictions;
  // generic LDS addr low 32 bits == LDS byte offset (ISA aperture rules)
  __builtin_amdgcn_global_load_async_to_lds_b128(
      (gbl_v4i_ptr)(uintptr_t)g, (lds_v4i_ptr)(uint32_t)(uintptr_t)l, 0, 0);
}
__device__ __forceinline__ void async_wait_all() {
  __builtin_amdgcn_s_wait_asynccnt(0);
}
#endif

// ---------------- types ----------------
typedef __attribute__((ext_vector_type(16))) __bf16 v16bf;
typedef __attribute__((ext_vector_type(8)))  __bf16 v8bf;
typedef __attribute__((ext_vector_type(8)))  float  v8f;

__device__ __forceinline__ uint16_t f2bf(float f) {
  uint32_t u = __builtin_bit_cast(uint32_t, f);
  uint32_t r = (u >> 16) & 1u;
  u += 0x7fffu + r;                       // round-to-nearest-even
  return (uint16_t)(u >> 16);
}

__device__ __forceinline__ v16bf cat16(v8bf lo, v8bf hi) {
  v16bf r;
#pragma unroll
  for (int i = 0; i < 8; ++i) { r[i] = lo[i]; r[i + 8] = hi[i]; }
  return r;
}

__device__ __forceinline__ v8f vzero8() {
  v8f z;
#pragma unroll
  for (int i = 0; i < 8; ++i) z[i] = 0.0f;
  return z;
}

__device__ __forceinline__ v8f wmma_bf16(v16bf a, v16bf b, v8f c) {
  return __builtin_amdgcn_wmma_f32_16x16x32_bf16(
      false, a, false, b, (short)0, c, false, false);
}

// ---------------- constants ----------------
#define BATCH 2
#define SEQ   2048
#define DIM   1024
#define HEADS 16
#define HDIM  64

#define TILE_M 128
#define TILE_N 128
#define TILE_K 32
#define LDSPAD 40   // GEMM LDS row length (elements): 80B -> conflict-free b128 reads
#define KPAD   72   // attn K tile row length: 144B stride
#define VTPAD  40   // attn V^T / P row length: 80B stride

// ---------------- fp32 -> bf16 convert ----------------
__global__ void cvt_kernel(const float* __restrict__ src,
                           uint16_t* __restrict__ dst, int n) {
  int i = (blockIdx.x * blockDim.x + threadIdx.x) * 4;
  if (i + 3 < n) {
    float4 v = *(const float4*)&src[i];
    ushort4 o;
    o.x = f2bf(v.x); o.y = f2bf(v.y); o.z = f2bf(v.z); o.w = f2bf(v.w);
    *(ushort4*)&dst[i] = o;
  } else {
    for (int j = i; j < n; ++j) dst[j] = f2bf(src[j]);
  }
}

// ---------------- mask -> bitmask pack ----------------
__global__ void pack_mask_kernel(const int* __restrict__ mask,
                                 unsigned int* __restrict__ out,
                                 int nwords, int wordsPerRow) {
  int w = blockIdx.x * blockDim.x + threadIdx.x;
  if (w >= nwords) return;
  int row = w / wordsPerRow;
  int cw  = w % wordsPerRow;
  const int* p = mask + (size_t)row * (wordsPerRow * 32) + cw * 32;
  unsigned int bits = 0;
#pragma unroll 4
  for (int j = 0; j < 32; ++j) bits |= (p[j] != 0 ? 1u : 0u) << j;
  out[w] = bits;
}

// ---------------- fused QKV GEMM (double-buffered) ----------------
// [4096,1024] x [1024,1024] + bias -> bf16 in head-major [B,H,S,HD]
__global__ __launch_bounds__(256) void gemm_qkv_kernel(
    const uint16_t* __restrict__ xb,
    const uint16_t* __restrict__ Wqb, const uint16_t* __restrict__ Wkb,
    const uint16_t* __restrict__ Wvb,
    const float* __restrict__ bq, const float* __restrict__ bk,
    const float* __restrict__ bv,
    uint16_t* __restrict__ Qb, uint16_t* __restrict__ Kb,
    uint16_t* __restrict__ Vb) {
  __shared__ uint16_t ldsA[2][TILE_M * LDSPAD];   // [row][k]
  __shared__ uint16_t ldsB[2][TILE_N * LDSPAD];   // [n][k] (transposed W tile)

  const int tid  = threadIdx.x;
  const int lane = tid & 31, wave = tid >> 5;
  const int half = lane >> 4, ln = lane & 15;
  const int wy = wave & 3, wx = wave >> 2;        // 4 waves along M, 2 along N
  const int m0 = blockIdx.y * TILE_M;
  const int n0 = blockIdx.x * TILE_N;

  const uint16_t* Wg  = (blockIdx.z == 0) ? Wqb : (blockIdx.z == 1) ? Wkb : Wvb;
  const float*    bia = (blockIdx.z == 0) ? bq  : (blockIdx.z == 1) ? bk  : bv;
  uint16_t*       dst = (blockIdx.z == 0) ? Qb  : (blockIdx.z == 1) ? Kb  : Vb;

  // staging coordinates
  int aRow[2], aCol[2];
#pragma unroll
  for (int p = 0; p < 2; ++p) {
    int e = tid * 8 + p * 2048;
    aRow[p] = e >> 5; aCol[p] = e & 31;
  }

  v8f acc[2][4];
#pragma unroll
  for (int i = 0; i < 2; ++i)
#pragma unroll
    for (int j = 0; j < 4; ++j) acc[i][j] = vzero8();

  const int KT = DIM / TILE_K;  // 32

  // ---- prologue: stage tile 0 into buffer 0 ----
#if USE_ASYNC_LDS
#pragma unroll
  for (int p = 0; p < 2; ++p)
    async_copy_b128(&xb[(m0 + aRow[p]) * DIM + aCol[p]],
                    &ldsA[0][aRow[p] * LDSPAD + aCol[p]]);
#else
  {
    uint4 ra[2];
#pragma unroll
    for (int p = 0; p < 2; ++p)
      ra[p] = *(const uint4*)&xb[(m0 + aRow[p]) * DIM + aCol[p]];
#pragma unroll
    for (int p = 0; p < 2; ++p)
      *(uint4*)&ldsA[0][aRow[p] * LDSPAD + aCol[p]] = ra[p];
  }
#endif
  {
    unsigned int rb[8];
#pragma unroll
    for (int i = 0; i < 8; ++i) {
      int idx = tid + i * 256;
      int k = idx >> 6, n2 = (idx & 63) * 2;
      rb[i] = *(const unsigned int*)&Wg[k * DIM + n0 + n2];
    }
#pragma unroll
    for (int i = 0; i < 8; ++i) {
      int idx = tid + i * 256;
      int k = idx >> 6, n2 = (idx & 63) * 2;
      ldsB[0][n2 * LDSPAD + k]       = (uint16_t)(rb[i] & 0xffffu);
      ldsB[0][(n2 + 1) * LDSPAD + k] = (uint16_t)(rb[i] >> 16);
    }
  }

  int cur = 0;
  for (int kt = 0; kt < KT; ++kt) {
#if USE_ASYNC_LDS
    async_wait_all();            // my async loads into ldsA[cur] are done
#endif
    __syncthreads();             // everyone's tile `cur` is visible
    const int nxt = cur ^ 1;

#if !USE_ASYNC_LDS
    uint4 ra[2];
#endif
    unsigned int rb[8];
    if (kt + 1 < KT) {
      const int kk = (kt + 1) * TILE_K;
#if USE_ASYNC_LDS
#pragma unroll
      for (int p = 0; p < 2; ++p)
        async_copy_b128(&xb[(m0 + aRow[p]) * DIM + kk + aCol[p]],
                        &ldsA[nxt][aRow[p] * LDSPAD + aCol[p]]);
#else
#pragma unroll
      for (int p = 0; p < 2; ++p)
        ra[p] = *(const uint4*)&xb[(m0 + aRow[p]) * DIM + kk + aCol[p]];
#endif
#pragma unroll
      for (int i = 0; i < 8; ++i) {
        int idx = tid + i * 256;
        int k = idx >> 6, n2 = (idx & 63) * 2;
        rb[i] = *(const unsigned int*)&Wg[(kk + k) * DIM + n0 + n2];
      }
    }

    // ---- compute on buffer `cur` ----
    v16bf afrag[2];
#pragma unroll
    for (int sm = 0; sm < 2; ++sm) {
      int row = wy * 32 + sm * 16 + ln;
      int kb  = half * 8;
      v8bf lo = *(const v8bf*)&ldsA[cur][row * LDSPAD + kb];
      v8bf hi = *(const v8bf*)&ldsA[cur][row * LDSPAD + kb + 16];
      afrag[sm] = cat16(lo, hi);
    }
#pragma unroll
    for (int sn = 0; sn < 4; ++sn) {
      int n   = wx * 64 + sn * 16 + ln;
      int kb2 = half * 8;
      v8bf lo = *(const v8bf*)&ldsB[cur][n * LDSPAD + kb2];
      v8bf hi = *(const v8bf*)&ldsB[cur][n * LDSPAD + kb2 + 16];
      v16bf bfrag = cat16(lo, hi);
#pragma unroll
      for (int sm = 0; sm < 2; ++sm)
        acc[sm][sn] = wmma_bf16(afrag[sm], bfrag, acc[sm][sn]);
    }

    // ---- stash tile kt+1 into buffer `nxt` ----
    if (kt + 1 < KT) {
#if !USE_ASYNC_LDS
#pragma unroll
      for (int p = 0; p < 2; ++p)
        *(uint4*)&ldsA[nxt][aRow[p] * LDSPAD + aCol[p]] = ra[p];
#endif
#pragma unroll
      for (int i = 0; i < 8; ++i) {
        int idx = tid + i * 256;
        int k = idx >> 6, n2 = (idx & 63) * 2;
        ldsB[nxt][n2 * LDSPAD + k]       = (uint16_t)(rb[i] & 0xffffu);
        ldsB[nxt][(n2 + 1) * LDSPAD + k] = (uint16_t)(rb[i] >> 16);
      }
    }
    cur = nxt;
  }

  // epilogue: bias + bf16 store into [B,H,S,HD]
#pragma unroll
  for (int sn = 0; sn < 4; ++sn) {
    int n  = n0 + wx * 64 + sn * 16 + ln;
    int h  = n >> 6, hd = n & 63;
    float bv_ = bia[n];
#pragma unroll
    for (int sm = 0; sm < 2; ++sm) {
#pragma unroll
      for (int r = 0; r < 8; ++r) {
        int m = m0 + wy * 32 + sm * 16 + half * 8 + r;
        int bb = m >> 11;            // m / SEQ
        int s  = m & (SEQ - 1);
        dst[((bb * HEADS + h) * SEQ + s) * HDIM + hd] =
            f2bf(acc[sm][sn][r] + bv_);
      }
    }
  }
}

// ---------------- flash attention per (b,h), double-buffered K/V ----------------
__global__ __launch_bounds__(128) void attn_kernel(
    const uint16_t* __restrict__ Qg, const uint16_t* __restrict__ Kg,
    const uint16_t* __restrict__ Vg, const unsigned int* __restrict__ mbits,
    const float* __restrict__ cmw, uint16_t* __restrict__ ctx) {
  __shared__ uint16_t ldsK[2][32 * KPAD];    // [key][hd], padded rows
  __shared__ uint16_t ldsVt[2][64 * VTPAD];  // [hd][key], padded rows
  __shared__ uint16_t ldsP[4][16 * VTPAD];   // per-wave P staging [m][k]

  const int tid  = threadIdx.x;
  const int lane = tid & 31, wave = tid >> 5;
  const int half = lane >> 4, ln = lane & 15;
  const int b = blockIdx.z, h = blockIdx.y;
  const int q0 = blockIdx.x * 64 + wave * 16;

  const uint16_t* Qbh = Qg + ((size_t)(b * HEADS + h) * SEQ) * HDIM;
  const uint16_t* Kbh = Kg + ((size_t)(b * HEADS + h) * SEQ) * HDIM;
  const uint16_t* Vbh = Vg + ((size_t)(b * HEADS + h) * SEQ) * HDIM;

  float mbias = 0.0f;
#pragma unroll
  for (int i = 0; i < 9; ++i) mbias += cmw[i];
  mbias *= (1.0f / 9.0f);
  const float scale = 0.125f;  // 1/sqrt(64)

  // Q A-fragments for hd 0..31 and 32..63 (held for whole kernel)
  v16bf qa[2];
  {
    const uint16_t* qrow = Qbh + (q0 + ln) * HDIM;
    int kb = half * 8;
#pragma unroll
    for (int f = 0; f < 2; ++f) {
      v8bf lo = *(const v8bf*)&qrow[f * 32 + kb];
      v8bf hi = *(const v8bf*)&qrow[f * 32 + kb + 16];
      qa[f] = cat16(lo, hi);
    }
  }

  // staging coordinates: K as 2 x b128, V as 16 scalars (transposed store)
  int kKr[2], kHd[2];
#pragma unroll
  for (int p = 0; p < 2; ++p) {
    int e = tid * 8 + p * 1024;
    kKr[p] = e >> 6; kHd[p] = e & 63;
  }

  float mrun[8], lrun[8];
  v8f O[4];
#pragma unroll
  for (int r = 0; r < 8; ++r) { mrun[r] = -3.0e38f; lrun[r] = 0.0f; }
#pragma unroll
  for (int t = 0; t < 4; ++t) O[t] = vzero8();

  const int NC = SEQ / 32;  // 64 chunks

  // ---- prologue: stage chunk 0 into buffer 0 ----
#if USE_ASYNC_LDS
#pragma unroll
  for (int p = 0; p < 2; ++p)
    async_copy_b128(&Kbh[kKr[p] * HDIM + kHd[p]],
                    &ldsK[0][kKr[p] * KPAD + kHd[p]]);
#else
  {
    uint4 rk[2];
#pragma unroll
    for (int p = 0; p < 2; ++p)
      rk[p] = *(const uint4*)&Kbh[kKr[p] * HDIM + kHd[p]];
#pragma unroll
    for (int p = 0; p < 2; ++p)
      *(uint4*)&ldsK[0][kKr[p] * KPAD + kHd[p]] = rk[p];
  }
#endif
  {
    uint16_t rv[16];
#pragma unroll
    for (int i = 0; i < 16; ++i) {
      int idx = tid + i * 128;
      rv[i] = Vbh[(idx >> 6) * HDIM + (idx & 63)];
    }
#pragma unroll
    for (int i = 0; i < 16; ++i) {
      int idx = tid + i * 128;
      ldsVt[0][(idx & 63) * VTPAD + (idx >> 6)] = rv[i];
    }
  }

  int cur = 0;
  for (int c = 0; c < NC; ++c) {
    const int key0 = c * 32;
#if USE_ASYNC_LDS
    async_wait_all();            // my K loads into ldsK[cur] are done
#endif
    __syncthreads();             // chunk `cur` fully visible
    const int nxt = cur ^ 1;

#if !USE_ASYNC_LDS
    uint4 rk[2];
#endif
    uint16_t rv[16];
    if (c + 1 < NC) {
      const int nk0 = key0 + 32;
#if USE_ASYNC_LDS
#pragma unroll
      for (int p = 0; p < 2; ++p)
        async_copy_b128(&Kbh[(nk0 + kKr[p]) * HDIM + kHd[p]],
                        &ldsK[nxt][kKr[p] * KPAD + kHd[p]]);
#else
#pragma unroll
      for (int p = 0; p < 2; ++p)
        rk[p] = *(const uint4*)&Kbh[(nk0 + kKr[p]) * HDIM + kHd[p]];
#endif
#pragma unroll
      for (int i = 0; i < 16; ++i) {
        int idx = tid + i * 128;
        rv[i] = Vbh[(nk0 + (idx >> 6)) * HDIM + (idx & 63)];
      }
    }

    // mask words for this 32-key chunk (one uint per row)
    unsigned int mw[8];
#pragma unroll
    for (int r = 0; r < 8; ++r) {
      int q = q0 + half * 8 + r;
      mw[r] = mbits[(b * SEQ + q) * (SEQ / 32) + c];
    }

    // ---- scores ----
    float val[2][8];
#pragma unroll
    for (int t = 0; t < 2; ++t) {
      v8f sc = vzero8();
#pragma unroll
      for (int f = 0; f < 2; ++f) {
        const uint16_t* kp =
            &ldsK[cur][(t * 16 + ln) * KPAD + f * 32 + half * 8];
        v8bf lo = *(const v8bf*)kp;
        v8bf hi = *(const v8bf*)(kp + 16);
        sc = wmma_bf16(qa[f], cat16(lo, hi), sc);
      }
#pragma unroll
      for (int r = 0; r < 8; ++r) {
        unsigned int bit = (mw[r] >> (t * 16 + ln)) & 1u;
        val[t][r] = bit ? (sc[r] * scale + mbias) : -1.0e9f;
      }
    }

    // ---- flash softmax update (row reductions within 16-lane halves) ----
    float corr[8];
#pragma unroll
    for (int r = 0; r < 8; ++r) {
      float vmax = fmaxf(val[0][r], val[1][r]);
      vmax = fmaxf(vmax, __shfl_xor(vmax, 1));
      vmax = fmaxf(vmax, __shfl_xor(vmax, 2));
      vmax = fmaxf(vmax, __shfl_xor(vmax, 4));
      vmax = fmaxf(vmax, __shfl_xor(vmax, 8));
      float mnew = fmaxf(mrun[r], vmax);
      corr[r] = __expf(mrun[r] - mnew);
      mrun[r] = mnew;
      float p0 = __expf(val[0][r] - mnew);
      float p1 = __expf(val[1][r] - mnew);
      float ps = p0 + p1;
      ps += __shfl_xor(ps, 1);
      ps += __shfl_xor(ps, 2);
      ps += __shfl_xor(ps, 4);
      ps += __shfl_xor(ps, 8);
      lrun[r] = lrun[r] * corr[r] + ps;
      int m = half * 8 + r;
      ldsP[wave][m * VTPAD + ln]      = f2bf(p0);
      ldsP[wave][m * VTPAD + 16 + ln] = f2bf(p1);
    }
#pragma unroll
    for (int ht = 0; ht < 4; ++ht)
#pragma unroll
      for (int r = 0; r < 8; ++r) O[ht][r] *= corr[r];

    // P as A-fragment (wave-local LDS restage)
    v16bf pa;
    {
      int kb = half * 8;
      v8bf lo = *(const v8bf*)&ldsP[wave][ln * VTPAD + kb];
      v8bf hi = *(const v8bf*)&ldsP[wave][ln * VTPAD + kb + 16];
      pa = cat16(lo, hi);
    }
    // O += P(16x32) x V(32x64)
#pragma unroll
    for (int ht = 0; ht < 4; ++ht) {
      const uint16_t* vp = &ldsVt[cur][(ht * 16 + ln) * VTPAD + half * 8];
      v8bf lo = *(const v8bf*)vp;
      v8bf hi = *(const v8bf*)(vp + 16);
      O[ht] = wmma_bf16(pa, cat16(lo, hi), O[ht]);
    }

    // ---- stash chunk c+1 into buffer `nxt` ----
    if (c + 1 < NC) {
#if !USE_ASYNC_LDS
#pragma unroll
      for (int p = 0; p < 2; ++p)
        *(uint4*)&ldsK[nxt][kKr[p] * KPAD + kHd[p]] = rk[p];
#endif
#pragma unroll
      for (int i = 0; i < 16; ++i) {
        int idx = tid + i * 128;
        ldsVt[nxt][(idx & 63) * VTPAD + (idx >> 6)] = rv[i];
      }
    }
    cur = nxt;
  }

  // normalize + store ctx in [B,S,D] (D = h*64+hd)
#pragma unroll
  for (int r = 0; r < 8; ++r) {
    float inv = 1.0f / lrun[r];
    int q = q0 + half * 8 + r;
#pragma unroll
    for (int ht = 0; ht < 4; ++ht) {
      int hd = ht * 16 + ln;
      ctx[((b * SEQ + q) * HEADS + h) * HDIM + hd] = f2bf(O[ht][r] * inv);
    }
  }
}

// ---------------- output GEMM (double-buffered): ctx x Wo + bo -> f32 ----------------
__global__ __launch_bounds__(256) void gemm_out_kernel(
    const uint16_t* __restrict__ ctxb, const uint16_t* __restrict__ Wob,
    const float* __restrict__ bo, float* __restrict__ out) {
  __shared__ uint16_t ldsA[2][TILE_M * LDSPAD];
  __shared__ uint16_t ldsB[2][TILE_N * LDSPAD];

  const int tid  = threadIdx.x;
  const int lane = tid & 31, wave = tid >> 5;
  const int half = lane >> 4, ln = lane & 15;
  const int wy = wave & 3, wx = wave >> 2;
  const int m0 = blockIdx.y * TILE_M;
  const int n0 = blockIdx.x * TILE_N;

  int aRow[2], aCol[2];
#pragma unroll
  for (int p = 0; p < 2; ++p) {
    int e = tid * 8 + p * 2048;
    aRow[p] = e >> 5; aCol[p] = e & 31;
  }

  v8f acc[2][4];
#pragma unroll
  for (int i = 0; i < 2; ++i)
#pragma unroll
    for (int j = 0; j < 4; ++j) acc[i][j] = vzero8();

  const int KT = DIM / TILE_K;

#if USE_ASYNC_LDS
#pragma unroll
  for (int p = 0; p < 2; ++p)
    async_copy_b128(&ctxb[(m0 + aRow[p]) * DIM + aCol[p]],
                    &ldsA[0][aRow[p] * LDSPAD + aCol[p]]);
#else
  {
    uint4 ra[2];
#pragma unroll
    for (int p = 0; p < 2; ++p)
      ra[p] = *(const uint4*)&ctxb[(m0 + aRow[p]) * DIM + aCol[p]];
#pragma unroll
    for (int p = 0; p < 2; ++p)
      *(uint4*)&ldsA[0][aRow[p] * LDSPAD + aCol[p]] = ra[p];
  }
#endif
  {
    unsigned int rb[8];
#pragma unroll
    for (int i = 0; i < 8; ++i) {
      int idx = tid + i * 256;
      int k = idx >> 6, n2 = (idx & 63) * 2;
      rb[i] = *(const unsigned int*)&Wob[k * DIM + n0 + n2];
    }
#pragma unroll
    for (int i = 0; i < 8; ++i) {
      int idx = tid + i * 256;
      int k = idx >> 6, n2 = (idx & 63) * 2;
      ldsB[0][n2 * LDSPAD + k]       = (uint16_t)(rb[i] & 0xffffu);
      ldsB[0][(n2 + 1) * LDSPAD + k] = (uint16_t)(rb[i] >> 16);
    }
  }

  int cur = 0;
  for (int kt = 0; kt < KT; ++kt) {
#if USE_ASYNC_LDS
    async_wait_all();
#endif
    __syncthreads();
    const int nxt = cur ^ 1;

#if !USE_ASYNC_LDS
    uint4 ra[2];
#endif
    unsigned int rb[8];
    if (kt + 1 < KT) {
      const int kk = (kt + 1) * TILE_K;
#if USE_ASYNC_LDS
#pragma unroll
      for (int p = 0; p < 2; ++p)
        async_copy_b128(&ctxb[(m0 + aRow[p]) * DIM + kk + aCol[p]],
                        &ldsA[nxt][aRow[p] * LDSPAD + aCol[p]]);
#else
#pragma unroll
      for (int p = 0; p < 2; ++p)
        ra[p] = *(const uint4*)&ctxb[(m0 + aRow[p]) * DIM + kk + aCol[p]];
#endif
#pragma unroll
      for (int i = 0; i < 8; ++i) {
        int idx = tid + i * 256;
        int k = idx >> 6, n2 = (idx & 63) * 2;
        rb[i] = *(const unsigned int*)&Wob[(kk + k) * DIM + n0 + n2];
      }
    }

    v16bf afrag[2];
#pragma unroll
    for (int sm = 0; sm < 2; ++sm) {
      int row = wy * 32 + sm * 16 + ln;
      int kb  = half * 8;
      v8bf lo = *(const v8bf*)&ldsA[cur][row * LDSPAD + kb];
      v8bf hi = *(const v8bf*)&ldsA[cur][row * LDSPAD + kb + 16];
      afrag[sm] = cat16(lo, hi);
    }
#pragma unroll
    for (int sn = 0; sn < 4; ++sn) {
      int n   = wx * 64 + sn * 16 + ln;
      int kb2 = half * 8;
      v8bf lo = *(const v8bf*)&ldsB[cur][n * LDSPAD + kb2];
      v8bf hi = *(const v8bf*)&ldsB[cur][n * LDSPAD + kb2 + 16];
      v16bf bfrag = cat16(lo, hi);
#pragma unroll
      for (int sm = 0; sm < 2; ++sm)
        acc[sm][sn] = wmma_bf16(afrag[sm], bfrag, acc[sm][sn]);
    }

    if (kt + 1 < KT) {
#if !USE_ASYNC_LDS
#pragma unroll
      for (int p = 0; p < 2; ++p)
        *(uint4*)&ldsA[nxt][aRow[p] * LDSPAD + aCol[p]] = ra[p];
#endif
#pragma unroll
      for (int i = 0; i < 8; ++i) {
        int idx = tid + i * 256;
        int k = idx >> 6, n2 = (idx & 63) * 2;
        ldsB[nxt][n2 * LDSPAD + k]       = (uint16_t)(rb[i] & 0xffffu);
        ldsB[nxt][(n2 + 1) * LDSPAD + k] = (uint16_t)(rb[i] >> 16);
      }
    }
    cur = nxt;
  }

#pragma unroll
  for (int sn = 0; sn < 4; ++sn) {
    int n = n0 + wx * 64 + sn * 16 + ln;
    float bv_ = bo[n];
#pragma unroll
    for (int sm = 0; sm < 2; ++sm) {
#pragma unroll
      for (int r = 0; r < 8; ++r) {
        int m = m0 + wy * 32 + sm * 16 + half * 8 + r;
        out[m * DIM + n] = acc[sm][sn][r] + bv_;
      }
    }
  }
}

// ---------------- host launcher ----------------
extern "C" void kernel_launch(void* const* d_in, const int* in_sizes, int n_in,
                              void* d_out, int out_size, void* d_ws,
                              size_t ws_size, hipStream_t stream) {
  const float* x    = (const float*)d_in[0];
  const float* Wq   = (const float*)d_in[1];
  const float* bq   = (const float*)d_in[2];
  const float* Wk   = (const float*)d_in[3];
  const float* bk   = (const float*)d_in[4];
  const float* Wv   = (const float*)d_in[5];
  const float* bv   = (const float*)d_in[6];
  const float* Wo   = (const float*)d_in[7];
  const float* bo   = (const float*)d_in[8];
  const float* cmw  = (const float*)d_in[9];
  const int*   mask = (const int*)d_in[10];
  float* out = (float*)d_out;

  const size_t MB = 1024 * 1024;
  uint8_t* ws = (uint8_t*)d_ws;
  uint16_t* xb   = (uint16_t*)(ws + 0);        // 8 MB
  uint16_t* Wqb  = (uint16_t*)(ws + 8 * MB);   // 2 MB
  uint16_t* Wkb  = (uint16_t*)(ws + 10 * MB);  // 2 MB
  uint16_t* Wvb  = (uint16_t*)(ws + 12 * MB);  // 2 MB
  uint16_t* Wob  = (uint16_t*)(ws + 14 * MB);  // 2 MB
  uint16_t* Qb   = (uint16_t*)(ws + 16 * MB);  // 8 MB
  uint16_t* Kb   = (uint16_t*)(ws + 24 * MB);  // 8 MB
  uint16_t* Vb   = (uint16_t*)(ws + 32 * MB);  // 8 MB
  uint16_t* ctxb = (uint16_t*)(ws + 40 * MB);  // 8 MB
  unsigned int* mbits = (unsigned int*)(ws + 48 * MB);  // 1 MB

  cvt_kernel<<<4096, 256, 0, stream>>>(x, xb, BATCH * SEQ * DIM);
  cvt_kernel<<<1024, 256, 0, stream>>>(Wq, Wqb, DIM * DIM);
  cvt_kernel<<<1024, 256, 0, stream>>>(Wk, Wkb, DIM * DIM);
  cvt_kernel<<<1024, 256, 0, stream>>>(Wv, Wvb, DIM * DIM);
  cvt_kernel<<<1024, 256, 0, stream>>>(Wo, Wob, DIM * DIM);

  const int nwords = BATCH * SEQ * (SEQ / 32);  // 262144
  pack_mask_kernel<<<(nwords + 255) / 256, 256, 0, stream>>>(mask, mbits,
                                                             nwords, SEQ / 32);

  gemm_qkv_kernel<<<dim3(DIM / TILE_N, (BATCH * SEQ) / TILE_M, 3), 256, 0,
                    stream>>>(xb, Wqb, Wkb, Wvb, bq, bk, bv, Qb, Kb, Vb);

  attn_kernel<<<dim3(SEQ / 64, HEADS, BATCH), 128, 0, stream>>>(
      Qb, Kb, Vb, mbits, cmw, ctxb);

  gemm_out_kernel<<<dim3(DIM / TILE_N, (BATCH * SEQ) / TILE_M), 256, 0,
                    stream>>>(ctxb, Wob, bo, out);
}
